// Net_43258910605835
// MI455X (gfx1250) — matrix operand order repaired and checked by reading, not compiled, not run on hardware
//
#include <hip/hip_runtime.h>
#include <hip/hip_bf16.h>

// ---------------------------------------------------------------------------
// CDNA5 / gfx1250 implementation of the light-field disparity network.
// Dominant compute (3x3x3 conv3d, 120ch, 36x48x48 vol) runs as implicit-GEMM
// on v_wmma_f32_16x16x32_f16.  Each 256-thread block owns a 64-voxel M tile
// staged in LDS; each of its 8 waves owns a 16-wide N tile and keeps 4
// independent accumulators (4 M-subtiles), so every B fragment load feeds 4
// independent WMMAs (no RAW chain, 4x arithmetic intensity on weights).
// ---------------------------------------------------------------------------

#define ANG   5
#define H2    240        // MacPI spatial
#define W2    240
#define NPIX  (H2*W2)    // 57600
#define HC    48
#define WC    48
#define DS3   36
#define NV3   (DS3*HC*WC)   // 82944 voxels of the cost volume
#define CP    128           // padded channel count (real = 120)
#define C3    120
#define TAPS  27
#define EPS   1e-5f
#define MT    64            // block M tile (voxels)

typedef __attribute__((ext_vector_type(16))) _Float16 v16h;
typedef __attribute__((ext_vector_type(8)))  float    v8f;

union AB16 { v16h v; uint4 q[2]; };

// ---------------------------------------------------------------------------
// SAI grid -> MacPI rearrange
__global__ void k_sai2macpi(const float* __restrict__ x, float* __restrict__ o) {
    int t = blockIdx.x * blockDim.x + threadIdx.x;
    if (t >= NPIX) return;
    int hh = t / W2, ww = t % W2;
    int h = hh / ANG, u = hh % ANG;
    int w = ww / ANG, v = ww % ANG;
    o[t] = x[(u * (H2/ANG) + h) * W2 + (v * (W2/ANG) + w)];
}

// ---------------------------------------------------------------------------
// Direct 3x3 conv2d, dilation=5, pad=5, on 240x240, fused BN / lrelu / resid.
__global__ void k_conv2d(const float* __restrict__ in, const float* __restrict__ w,
                         const float* __restrict__ g, const float* __restrict__ b,
                         const float* __restrict__ m, const float* __restrict__ vv,
                         const float* __restrict__ resid, float* __restrict__ out,
                         int Cin, int Cout, int relu) {
    int t = blockIdx.x * blockDim.x + threadIdx.x;
    if (t >= Cout * NPIX) return;
    int co  = t / NPIX;
    int pix = t % NPIX;
    int y = pix / W2, x = pix % W2;
    float acc = 0.f;
    for (int ci = 0; ci < Cin; ++ci) {
        const float* ip = in + ci * NPIX;
        const float* wp = w + (co * Cin + ci) * 9;
        #pragma unroll
        for (int ky = 0; ky < 3; ++ky) {
            int sy = y + (ky - 1) * ANG;
            if (sy < 0 || sy >= H2) continue;
            #pragma unroll
            for (int kx = 0; kx < 3; ++kx) {
                int sx = x + (kx - 1) * ANG;
                if (sx < 0 || sx >= W2) continue;
                acc += ip[sy * W2 + sx] * wp[ky * 3 + kx];
            }
        }
    }
    if (g) {
        float inv = rsqrtf(vv[co] + EPS);
        acc = (acc - m[co]) * inv * g[co] + b[co];
    }
    if (relu) acc = acc > 0.f ? acc : 0.1f * acc;
    if (resid) acc += resid[t];
    out[t] = acc;
}

// ---------------------------------------------------------------------------
// build_cost: stride-5 dilated 5x5 conv for each integer disparity (grid.z).
__constant__ int c_dil[5] = {11, 6, 1, 4, 9};   // d = -2..2
__constant__ int c_pad[5] = {20, 10, 0, 6, 16};

__global__ void k_build_cost(const float* __restrict__ feat, const float* __restrict__ wd,
                             float* __restrict__ int_cost) {
    int t = blockIdx.x * blockDim.x + threadIdx.x;
    if (t >= C3 * HC * WC) return;
    int di  = blockIdx.z;                 // 0..4
    int dil = c_dil[di], pad = c_pad[di];
    int co = t / (HC * WC);
    int p  = t % (HC * WC);
    int oy = p / WC, ox = p % WC;
    float acc = 0.f;
    for (int ci = 0; ci < 8; ++ci) {
        const float* ip = feat + ci * NPIX;
        const float* wp = wd + (co * 8 + ci) * 25;
        #pragma unroll
        for (int ky = 0; ky < 5; ++ky) {
            int sy = oy * ANG - pad + ky * dil;
            if (sy < 0 || sy >= H2) continue;
            #pragma unroll
            for (int kx = 0; kx < 5; ++kx) {
                int sx = ox * ANG - pad + kx * dil;
                if (sx < 0 || sx >= W2) continue;
                acc += ip[sy * W2 + sx] * wp[ky * 5 + kx];
            }
        }
    }
    int_cost[(di * C3 + co) * (HC * WC) + p] = acc;
}

// ---------------------------------------------------------------------------
// Blend integer-disparity costs into the 36-slice volume and pack to f16
// channels-last [NV3][CP].  (Integer-d slices get weight 0 — faithful.)
__global__ void k_blend_pack(const float* __restrict__ int_cost, _Float16* __restrict__ v0) {
    long long t = (long long)blockIdx.x * blockDim.x + threadIdx.x;
    if (t >= (long long)NV3 * CP) return;
    int vi = (int)(t / CP);
    int c  = (int)(t % CP);
    int dsl = vi / (HC * WC);        // 0..35
    int p   = vi % (HC * WC);
    int j   = dsl / 4;               // 0..8   d = -2 + 0.5j
    float val = 0.f;
    if (c < C3) {
        int dl = j >> 1, dh = (j + 1) >> 1;
        float wgt = (j & 1) ? 0.5f : 0.0f;     // both weights, faithful to ref
        val = wgt * int_cost[(dl * C3 + c) * (HC * WC) + p]
            + wgt * int_cost[(dh * C3 + c) * (HC * WC) + p];
    }
    v0[(long long)vi * CP + c] = (_Float16)val;
}

// ---------------------------------------------------------------------------
// Pack conv3d weights f32 [Cout][Cin][3][3][3] -> f16 [tap][128][128]
__global__ void k_pack_w3d(const float* __restrict__ w, _Float16* __restrict__ wp,
                           int Cout, int Cin) {
    int t = blockIdx.x * blockDim.x + threadIdx.x;
    if (t >= TAPS * CP * CP) return;
    int tap = t / (CP * CP);
    int r   = t % (CP * CP);
    int co = r / CP, ci = r % CP;
    float val = 0.f;
    if (co < Cout && ci < Cin) val = w[(co * Cin + ci) * TAPS + tap];
    wp[t] = (_Float16)val;
}

// ---------------------------------------------------------------------------
// Implicit-GEMM conv3d, 3x3x3 pad 1, on [36][48][48] volume, channels-last f16
// padded to 128.  Block = 256 threads = 8 waves; M tile = 64 voxels in LDS;
// each wave owns a 16-wide N tile with 4 M-subtile accumulators.  Per tap a
// wave issues 4 K-chunks x 4 independent WMMAs = 16 v_wmma_f32_16x16x32_f16.
// Fused epilogue: BN, leaky-relu, residual add; writes f16 packed and/or f32.
#define F_RELU 1

__global__ __launch_bounds__(256)
void k_conv3d_wmma(const _Float16* __restrict__ act, const _Float16* __restrict__ wp,
                   const float* __restrict__ g, const float* __restrict__ b,
                   const float* __restrict__ m, const float* __restrict__ vv,
                   const _Float16* __restrict__ resid,
                   _Float16* __restrict__ out16, float* __restrict__ out32,
                   int CoutReal, int flags) {
    __shared__ __attribute__((aligned(16))) _Float16 sA[MT * CP];   // 16 KB

    const int lane = threadIdx.x & 31;
    const int wv   = threadIdx.x >> 5;          // 0..7  -> N tile
    const int mb   = blockIdx.x * MT;           // base voxel of block M tile

    // staging: each thread fills 4 rows (16B chunk each) of the 64-row tile
    const int smi  = threadIdx.x >> 4;          // 0..15 base voxel row
    const int sseg = threadIdx.x & 15;          // 0..15 -> 8 halves each
    int sd0[4], sy0[4], sx0[4];
    #pragma unroll
    for (int s = 0; s < 4; ++s) {
        int vox = mb + smi + 16 * s;
        sd0[s] = vox / (HC * WC);
        int sp = vox % (HC * WC);
        sy0[s] = sp / WC;
        sx0[s] = sp % WC;
    }

    const int Mrow = lane & 15;
    const int aoff = (lane < 16) ? 0 : 8;       // A-fragment half offset
    const int boff = (lane < 16) ? 0 : 8;       // B-fragment half offset
    const int ncol = wv * 16 + (lane & 15);     // output channel (padded)

    v8f acc[4] = {{}, {}, {}, {}};

    for (int tap = 0; tap < TAPS; ++tap) {
        const int kd = tap / 9 - 1;
        const int kr = tap % 9;
        const int kh = kr / 3 - 1;
        const int kw = kr % 3 - 1;

        __syncthreads();   // protect previous tap's reads
        #pragma unroll
        for (int s = 0; s < 4; ++s) {
            int dd = sd0[s] + kd, yy = sy0[s] + kh, xx = sx0[s] + kw;
            uint4 val = make_uint4(0u, 0u, 0u, 0u);
            if (dd >= 0 && dd < DS3 && yy >= 0 && yy < HC && xx >= 0 && xx < WC) {
                val = *(const uint4*)(act +
                        ((long long)(dd * (HC * WC) + yy * WC + xx) * CP + sseg * 8));
            }
            *(uint4*)(sA + (smi + 16 * s) * CP + sseg * 8) = val;
        }
        __syncthreads();

        const _Float16* wbase = wp + ((long long)(tap * CP + ncol)) * CP;
        __builtin_prefetch(wbase + CP * CP, 0, 1);   // next tap's weights

        #pragma unroll
        for (int kc = 0; kc < 4; ++kc) {
            // B 32x16 f16 fragment from global (weights, L2 resident) — loaded
            // once per K chunk, feeds 4 independent WMMAs below.
            AB16 bf;
            const _Float16* bp = wbase + kc * 32 + boff;
            bf.q[0] = *(const uint4*)(bp);
            bf.q[1] = *(const uint4*)(bp + 16);
            #pragma unroll
            for (int ms = 0; ms < 4; ++ms) {
                AB16 a;   // A 16x32 f16 fragment from LDS (ds_load_b128 x2)
                const _Float16* ap = sA + (ms * 16 + Mrow) * CP + kc * 32 + aoff;
                a.q[0] = *(const uint4*)(ap);
                a.q[1] = *(const uint4*)(ap + 16);
                acc[ms] = __builtin_amdgcn_wmma_f32_16x16x32_f16(
                        false, a.v, false, bf.v, (short)0, acc[ms], false, false);
            }
        }
    }

    // -------- epilogue: BN + lrelu + residual, write f16 packed / f32 ------
    const int nc = (ncol < CoutReal) ? ncol : 0;
    float sc = 1.f, sh = 0.f;
    if (g) {
        float inv = rsqrtf(vv[nc] + EPS);
        sc = g[nc] * inv;
        sh = b[nc] - m[nc] * sc;
    }
    const int mofs = (lane >= 16) ? 8 : 0;
    #pragma unroll
    for (int ms = 0; ms < 4; ++ms) {
        #pragma unroll
        for (int r = 0; r < 8; ++r) {
            int vox = mb + ms * 16 + r + mofs;
            float val = acc[ms][r];
            if (g) val = val * sc + sh;
            if (flags & F_RELU) val = val > 0.f ? val : 0.1f * val;
            if (resid) val += (float)resid[(long long)vox * CP + ncol];
            if (out16)
                out16[(long long)vox * CP + ncol] =
                    (_Float16)((ncol < CoutReal) ? val : 0.f);
            if (out32 && ncol < CoutReal)
                out32[(long long)ncol * NV3 + vox] = val;
        }
    }
}

// ---------------------------------------------------------------------------
// concat_blocks pixel shuffle: [4][36][48][48] -> [4][9][96][96]
__global__ void k_pixel_shuffle(const float* __restrict__ in, float* __restrict__ out) {
    int t = blockIdx.x * blockDim.x + threadIdx.x;
    if (t >= 4 * 9 * 96 * 96) return;
    int c  = t / (9 * 96 * 96);
    int r  = t % (9 * 96 * 96);
    int dd = r / (96 * 96);
    int p  = r % (96 * 96);
    int Y = p / 96, X = p % 96;
    int h = Y >> 1, s1 = Y & 1;
    int w = X >> 1, s2 = X & 1;
    out[t] = in[((c * DS3 + dd * 4 + s1 * 2 + s2) * (HC * WC)) + h * WC + w];
}

// ---------------------------------------------------------------------------
// Small direct conv3d (Cin,Cout <= 4) on [C][9][96][96], pad 1, fused BN/lrelu.
__global__ void k_conv3d_small(const float* __restrict__ in, const float* __restrict__ w,
                               const float* __restrict__ g, const float* __restrict__ b,
                               const float* __restrict__ m, const float* __restrict__ vv,
                               float* __restrict__ out, int Cin, int Cout, int relu) {
    int t = blockIdx.x * blockDim.x + threadIdx.x;
    int NV = 9 * 96 * 96;
    if (t >= Cout * NV) return;
    int co = t / NV;
    int p  = t % NV;
    int d = p / (96 * 96);
    int r = p % (96 * 96);
    int y = r / 96, x = r % 96;
    float acc = 0.f;
    for (int ci = 0; ci < Cin; ++ci) {
        #pragma unroll
        for (int kd = -1; kd <= 1; ++kd) {
            int sd = d + kd;
            if (sd < 0 || sd >= 9) continue;
            #pragma unroll
            for (int ky = -1; ky <= 1; ++ky) {
                int sy = y + ky;
                if (sy < 0 || sy >= 96) continue;
                #pragma unroll
                for (int kx = -1; kx <= 1; ++kx) {
                    int sx = x + kx;
                    if (sx < 0 || sx >= 96) continue;
                    acc += in[((ci * 9 + sd) * 96 + sy) * 96 + sx] *
                           w[(((co * Cin + ci) * 3 + kd + 1) * 3 + ky + 1) * 3 + kx + 1];
                }
            }
        }
    }
    if (g) {
        float inv = rsqrtf(vv[co] + EPS);
        acc = (acc - m[co]) * inv * g[co] + b[co];
    }
    if (relu) acc = acc > 0.f ? acc : 0.1f * acc;
    out[t] = acc;
}

// ---------------------------------------------------------------------------
// Softmax-over-9 regression -> super_disp [96][96]
__global__ void k_regression(const float* __restrict__ cost, float* __restrict__ out) {
    int t = blockIdx.x * blockDim.x + threadIdx.x;
    if (t >= 96 * 96) return;
    float c[9], mx = -1e30f;
    #pragma unroll
    for (int d = 0; d < 9; ++d) { c[d] = cost[d * 96 * 96 + t]; mx = fmaxf(mx, c[d]); }
    float s = 0.f;
    #pragma unroll
    for (int d = 0; d < 9; ++d) { c[d] = __expf(c[d] - mx); s += c[d]; }
    float inv = 1.f / s, disp = 0.f;
    #pragma unroll
    for (int d = 0; d < 9; ++d) disp += c[d] * inv * (float)(d - 4);
    out[t] = disp;
}

// ---------------------------------------------------------------------------
// Keys cubic (a = -0.5) 2x downsample 96->48, then /2
__global__ void k_cubic_down(const float* __restrict__ in, float* __restrict__ out) {
    int t = blockIdx.x * blockDim.x + threadIdx.x;
    if (t >= 48 * 48) return;
    int oy = t / 48, ox = t % 48;
    // src = 2*o + 0.5 ; taps at 2o-1..2o+2 with fixed weights for frac 0.5
    const float wk[4] = {-0.0625f, 0.5625f, 0.5625f, -0.0625f};
    float acc = 0.f;
    #pragma unroll
    for (int i = 0; i < 4; ++i) {
        int sy = 2 * oy - 1 + i;
        sy = sy < 0 ? 0 : (sy > 95 ? 95 : sy);
        #pragma unroll
        for (int j = 0; j < 4; ++j) {
            int sx = 2 * ox - 1 + j;
            sx = sx < 0 ? 0 : (sx > 95 ? 95 : sx);
            acc += wk[i] * wk[j] * in[sy * 96 + sx];
        }
    }
    out[t] = acc * 0.5f;   // / SCALE_T
}

// ---------------------------------------------------------------------------
// Host sequencing
// ---------------------------------------------------------------------------
static inline size_t alignup(size_t x) { return (x + 255) & ~(size_t)255; }

extern "C" void kernel_launch(void* const* d_in, const int* in_sizes, int n_in,
                              void* d_out, int out_size, void* d_ws, size_t ws_size,
                              hipStream_t stream) {
    (void)in_sizes; (void)n_in; (void)out_size; (void)ws_size;
    auto F = [&](int i) { return (const float*)d_in[i]; };

    // -------- input index map (setup_inputs dict insertion order, DFS) -----
    // 0: x
    // init: conv1=1, bn1(g,b,m,v)=2..5, conv2=6, bn2=7..10,
    //       res[i] (i=0..7): conv1=11+10i, bn1=12+10i.., conv2=16+10i, bn2=17+10i..,
    //       conv3=91, bn3=92..95, conv4=96
    // dsafe=97
    // agg: conv1=98, bn1=99..102,
    //      res0: conv1=103, bn1=104..107, conv2=108, bn2=109..112
    //      res1: conv1=113, bn1=114..117, conv2=118, bn2=119..122
    //      conv2=123
    // agg2: conv1=124, bn1=125..128, conv2=129
    const float* X = F(0);

    // -------- workspace carve ---------------------------------------------
    char* ws = (char*)d_ws;
    size_t off = 0;
    auto take = [&](size_t bytes) { char* p = ws + off; off += alignup(bytes); return p; };
    float*    macpi   = (float*)take(NPIX * 4);
    float*    featA   = (float*)take(16 * NPIX * 4);
    float*    featB   = (float*)take(16 * NPIX * 4);
    float*    feat8   = (float*)take(8 * NPIX * 4);
    float*    icost   = (float*)take((size_t)5 * C3 * HC * WC * 4);
    _Float16* V0      = (_Float16*)take((size_t)NV3 * CP * 2);
    _Float16* V1      = (_Float16*)take((size_t)NV3 * CP * 2);
    _Float16* V2      = (_Float16*)take((size_t)NV3 * CP * 2);
    _Float16* wpk[6];
    for (int i = 0; i < 6; ++i) wpk[i] = (_Float16*)take((size_t)TAPS * CP * CP * 2);
    float*    aggout  = (float*)take((size_t)4 * NV3 * 4);
    float*    shuf    = (float*)take((size_t)4 * 9 * 96 * 96 * 4);
    float*    g1      = (float*)take((size_t)4 * 9 * 96 * 96 * 4);
    float*    g2      = (float*)take((size_t)9 * 96 * 96 * 4);

    dim3 B(256);
    auto gr = [&](long long n) { return dim3((unsigned)((n + 255) / 256)); };

    // -------- init 2D feature stage ---------------------------------------
    hipLaunchKernelGGL(k_sai2macpi, gr(NPIX), B, 0, stream, X, macpi);
    // conv1: 1->16, bn1, lrelu
    hipLaunchKernelGGL(k_conv2d, gr(16LL * NPIX), B, 0, stream,
                       macpi, F(1), F(2), F(3), F(4), F(5), nullptr, featA, 1, 16, 1);
    // conv2: 16->16, bn2, lrelu
    hipLaunchKernelGGL(k_conv2d, gr(16LL * NPIX), B, 0, stream,
                       featA, F(6), F(7), F(8), F(9), F(10), nullptr, featB, 16, 16, 1);
    // 8 residual blocks:  featB -> featA (relu) -> featB (+resid, in-place ok)
    for (int i = 0; i < 8; ++i) {
        int base = 11 + 10 * i;
        hipLaunchKernelGGL(k_conv2d, gr(16LL * NPIX), B, 0, stream,
                           featB, F(base), F(base+1), F(base+2), F(base+3), F(base+4),
                           nullptr, featA, 16, 16, 1);
        hipLaunchKernelGGL(k_conv2d, gr(16LL * NPIX), B, 0, stream,
                           featA, F(base+5), F(base+6), F(base+7), F(base+8), F(base+9),
                           featB, featB, 16, 16, 0);
    }
    // conv3: 16->16, bn3, lrelu
    hipLaunchKernelGGL(k_conv2d, gr(16LL * NPIX), B, 0, stream,
                       featB, F(91), F(92), F(93), F(94), F(95), nullptr, featA, 16, 16, 1);
    // conv4: 16->8, no bn/relu
    hipLaunchKernelGGL(k_conv2d, gr(8LL * NPIX), B, 0, stream,
                       featA, F(96), nullptr, nullptr, nullptr, nullptr, nullptr,
                       feat8, 16, 8, 0);

    // -------- cost volume --------------------------------------------------
    {
        dim3 g((unsigned)((C3 * HC * WC + 255) / 256), 1, 5);
        hipLaunchKernelGGL(k_build_cost, g, B, 0, stream, feat8, F(97), icost);
    }
    hipLaunchKernelGGL(k_blend_pack, gr((long long)NV3 * CP), B, 0, stream, icost, V0);

    // -------- pack conv3d weights to f16 [tap][coutP][cinP] ----------------
    const int widx[6]  = {98, 103, 108, 113, 118, 123};
    const int wcout[6] = {C3, C3, C3, C3, C3, 4};
    for (int i = 0; i < 6; ++i)
        hipLaunchKernelGGL(k_pack_w3d, gr((long long)TAPS * CP * CP), B, 0, stream,
                           F(widx[i]), wpk[i], wcout[i], C3);

    // -------- WMMA conv3d chain (the heavy part) ---------------------------
    dim3 G3((unsigned)(NV3 / MT));   // 1296 blocks, 256 threads, 8 waves
    // agg.conv1: V0 -> V1 (bn1, lrelu)
    hipLaunchKernelGGL(k_conv3d_wmma, G3, B, 0, stream,
                       V0, wpk[0], F(99), F(100), F(101), F(102),
                       (const _Float16*)nullptr, V1, (float*)nullptr, C3, F_RELU);
    // res0.conv1: V1 -> V2 (bn, lrelu)
    hipLaunchKernelGGL(k_conv3d_wmma, G3, B, 0, stream,
                       V1, wpk[1], F(104), F(105), F(106), F(107),
                       (const _Float16*)nullptr, V2, (float*)nullptr, C3, F_RELU);
    // res0.conv2: V2 -> V0  (bn, +V1 residual, no relu)
    hipLaunchKernelGGL(k_conv3d_wmma, G3, B, 0, stream,
                       V2, wpk[2], F(109), F(110), F(111), F(112),
                       V1, V0, (float*)nullptr, C3, 0);
    // res1.conv1: V0 -> V2 (bn, lrelu)
    hipLaunchKernelGGL(k_conv3d_wmma, G3, B, 0, stream,
                       V0, wpk[3], F(114), F(115), F(116), F(117),
                       (const _Float16*)nullptr, V2, (float*)nullptr, C3, F_RELU);
    // res1.conv2: V2 -> V1 (bn, +V0 residual, no relu)
    hipLaunchKernelGGL(k_conv3d_wmma, G3, B, 0, stream,
                       V2, wpk[4], F(119), F(120), F(121), F(122),
                       V0, V1, (float*)nullptr, C3, 0);
    // agg.conv2: V1 -> aggout f32 [4][36][48][48] (no bn/relu)
    hipLaunchKernelGGL(k_conv3d_wmma, G3, B, 0, stream,
                       V1, wpk[5], (const float*)nullptr, (const float*)nullptr,
                       (const float*)nullptr, (const float*)nullptr,
                       (const _Float16*)nullptr, (_Float16*)nullptr, aggout, 4, 0);

    // -------- upsample & second aggregation --------------------------------
    hipLaunchKernelGGL(k_pixel_shuffle, gr(4LL * 9 * 96 * 96), B, 0, stream, aggout, shuf);
    hipLaunchKernelGGL(k_conv3d_small, gr(4LL * 9 * 96 * 96), B, 0, stream,
                       shuf, F(124), F(125), F(126), F(127), F(128), g1, 4, 4, 1);
    hipLaunchKernelGGL(k_conv3d_small, gr(1LL * 9 * 96 * 96), B, 0, stream,
                       g1, F(129), nullptr, nullptr, nullptr, nullptr, g2, 4, 1, 0);

    // -------- regression + cubic downsample --------------------------------
    float* out = (float*)d_out;             // [down 48*48 | super 96*96]
    float* super_disp = out + 48 * 48;
    hipLaunchKernelGGL(k_regression, gr(96LL * 96), B, 0, stream, g2, super_disp);
    hipLaunchKernelGGL(k_cubic_down, gr(48LL * 48), B, 0, stream, super_disp, out);
}
